// MotionWKV_TokenMix_44684839747664
// MI455X (gfx1250) — matrix operand order repaired and checked by reading, not compiled, not run on hardware
//
#include <hip/hip_runtime.h>

#define C_DIM 1024
#define B_DIM 8
#define T_DIM 2048
#define M_DIM (B_DIM * T_DIM)   // 16384
#define NCHUNK 32
#define LCHUNK (T_DIM / NCHUNK) // 64
#define NKSTEP (C_DIM / 32)     // 32 K-iterations

// Fused-kernel LDS geometry (u32 units per buffer):
//   B tile p (64x32 bf16)  @ p*1024   (3 tiles = 3072 u32 = 12 KB)
//   A tile p (128x32 bf16) @ 3072 + p*2048 (3 tiles = 6144 u32 = 24 KB)
// per-buffer = 9216 u32 (36 KB), two buffers = 72 KB.
#define KVR_BUF_U32 9216
#define KVR_BUF_BYTES 36864

typedef __bf16 v16bf __attribute__((ext_vector_type(16)));
typedef float  v8f   __attribute__((ext_vector_type(8)));
typedef unsigned int u32x4 __attribute__((ext_vector_type(4)));
typedef int          i32x4 __attribute__((ext_vector_type(4)));
typedef int          i32x8 __attribute__((ext_vector_type(8)));

union FragAB { v16bf v; unsigned u[8]; };

#ifndef __has_builtin
#define __has_builtin(x) 0
#endif
#if __has_builtin(__builtin_amdgcn_tensor_load_to_lds) && \
    __has_builtin(__builtin_amdgcn_s_wait_tensorcnt)
#define HAVE_TDM 1
#else
#define HAVE_TDM 0
#endif
#if HAVE_TDM && __has_include(<hip/amd_detail/amd_gfx1250_TDM.h>)
#define TDM_6ARG 1
#else
#define TDM_6ARG 0
#endif

__device__ __forceinline__ unsigned pack_bf2(float lo, float hi) {
    __bf16 l = (__bf16)lo;
    __bf16 h = (__bf16)hi;
    unsigned short ul = __builtin_bit_cast(unsigned short, l);
    unsigned short uh = __builtin_bit_cast(unsigned short, h);
    return (unsigned)ul | ((unsigned)uh << 16);
}

#if HAVE_TDM
// TDM 2D tile load: `rows` x 32 bf16 cols from a row-major matrix with row
// stride C_DIM elements, into LDS at byte offset lds_off.
// D# layout per CDNA5 ISA (2D: groups 2/3 zero).
__device__ __forceinline__ void tdm_load_tile(const unsigned short* gptr,
                                              unsigned lds_off, int rows) {
    unsigned long long ga = (unsigned long long)(__SIZE_TYPE__)gptr;
    u32x4 g0;
    g0[0] = 1u;                                   // count=1 valid, is_restore=0
    g0[1] = lds_off;                              // lds_addr (bytes)
    g0[2] = (unsigned)(ga & 0xFFFFFFFFu);         // global_addr[31:0]
    g0[3] = (unsigned)((ga >> 32) & 0x01FFFFFFu)  // global_addr[56:32]
            | (2u << 30);                         // type=2 (image)
    i32x8 g1;
    g1[0] = (1 << 16);                            // data_size=1 (2B), wg_mask=0
    g1[1] = (int)((C_DIM & 0xFFFF) << 16);        // tensor_dim0 low16 @bit48
    g1[2] = (int)(((C_DIM >> 16) & 0xFFFF) |      // tensor_dim0 hi16
                  ((C_DIM & 0xFFFF) << 16));      // tensor_dim1 low16
    g1[3] = (int)(((C_DIM >> 16) & 0xFFFF) |      // tensor_dim1 hi16
                  (32u << 16));                   // tile_dim0 = 32
    g1[4] = rows;                                 // tile_dim1=rows, tile_dim2=0
    g1[5] = C_DIM;                                // tensor_dim0_stride low32
    g1[6] = 0;
    g1[7] = 0;
    i32x4 z4 = {0, 0, 0, 0};
#if TDM_6ARG
    i32x8 z8 = {0, 0, 0, 0, 0, 0, 0, 0};
    __builtin_amdgcn_tensor_load_to_lds(g0, g1, z4, z4, z8, 0);
#else
    __builtin_amdgcn_tensor_load_to_lds(g0, g1, z4, z4, 0);
#endif
}
#endif

// ---------------------------------------------------------------------------
// fp32 -> bf16 conversion of the 4 weight matrices into workspace
// ---------------------------------------------------------------------------
__global__ void cvt_w_kernel(const float* __restrict__ w0, const float* __restrict__ w1,
                             const float* __restrict__ w2, const float* __restrict__ w3,
                             unsigned short* __restrict__ dst) {
    int mat = blockIdx.y;
    const float* s = (mat == 0) ? w0 : (mat == 1) ? w1 : (mat == 2) ? w2 : w3;
    size_t i = ((size_t)blockIdx.x * blockDim.x + threadIdx.x) * 4;
    float4 f = *(const float4*)(s + i);
    unsigned short* d = dst + (size_t)mat * C_DIM * C_DIM + i;
    uint2 p;
    p.x = pack_bf2(f.x, f.y);
    p.y = pack_bf2(f.z, f.w);
    *(uint2*)d = p;
}

// ---------------------------------------------------------------------------
// Fused k/v/r projection GEMM (einsum btc,dc->btd for 3 weight matrices),
// temporal-shift mix fused into the A staging; x is loaded ONCE for all three.
// Block = 256 threads (8 waves). WG tile = 128(M) x 64(N); each wave computes
// 32x32 per projection (acc = 3*2*2 v8f = 96 VGPRs -> no accumulator spills).
// Double-buffered LDS (72 KB); TDM for K-step i+1 overlaps compute of step i.
// ---------------------------------------------------------------------------
__global__ void gemm_kvr_kernel(const float* __restrict__ x,        // (M,C) f32
                                const float* __restrict__ mk,
                                const float* __restrict__ mv,
                                const float* __restrict__ mr,       // (C,)
                                const unsigned short* __restrict__ Wkb,
                                const unsigned short* __restrict__ Wvb,
                                const unsigned short* __restrict__ Wrb, // (C,C) bf16
                                float* __restrict__ ko,
                                float* __restrict__ vo,
                                float* __restrict__ ro) {           // (M,C) f32
    extern __shared__ char smem[];
    unsigned* su = (unsigned*)smem;

    const int tid  = threadIdx.x;
    const int n0   = blockIdx.x * 64;
    const int m0   = blockIdx.y * 128;
    const int lane = tid & 31;
    const int wave = tid >> 5;
    const int waveM = (wave & 3) << 5;   // 0,32,64,96
    const int waveN = (wave >> 2) << 5;  // 0,32
    const int m16  = lane & 15;
    const int g    = lane >> 4;

    const unsigned short* Wp[3] = {Wkb, Wvb, Wrb};

    v8f acc[3][2][2];
#pragma unroll
    for (int p = 0; p < 3; ++p)
#pragma unroll
        for (int i = 0; i < 2; ++i)
#pragma unroll
            for (int j = 0; j < 2; ++j) acc[p][i][j] = {};

    // ---- staging helpers -------------------------------------------------
    auto stage_B = [&](int buf, int k0) {
#if HAVE_TDM
        if (wave == 0) {
#pragma unroll
            for (int p = 0; p < 3; ++p)
                tdm_load_tile(Wp[p] + (size_t)n0 * C_DIM + k0,
                              (unsigned)(buf * KVR_BUF_BYTES + p * 4096), 64);
        }
#else
        // 64 rows x 32 bf16 per projection; 1 uint4 per thread per projection
        const int row = tid >> 2;      // 0..63
        const int qt  = tid & 3;       // 0..3
#pragma unroll
        for (int p = 0; p < 3; ++p) {
            const uint4* src = (const uint4*)(Wp[p] + (size_t)(n0 + row) * C_DIM + k0);
            uint4* dst = (uint4*)(su + buf * KVR_BUF_U32 + p * 1024) + row * 4;
            dst[qt] = src[qt];
        }
#endif
        (void)buf; (void)k0;
    };

    auto stage_A = [&](int buf, int k0) {
        int row = tid >> 3;            // 0..31
        const int kq = (tid & 7) * 4;  // 0..28 step 4
        const float4 x0 = *(const float4*)(mk + k0 + kq);
        const float4 x1 = *(const float4*)(mv + k0 + kq);
        const float4 x2 = *(const float4*)(mr + k0 + kq);
        unsigned* Ab = su + buf * KVR_BUF_U32 + 3072;
#pragma unroll
        for (int it = 0; it < 4; ++it, row += 32) {
            const int m = m0 + row;
            const float4 xc = *(const float4*)(x + (size_t)m * C_DIM + k0 + kq);
            float4 xp = make_float4(0.f, 0.f, 0.f, 0.f);
            if ((m % T_DIM) != 0)      // temporal shift: zero at t==0
                xp = *(const float4*)(x + (size_t)(m - 1) * C_DIM + k0 + kq);
            const float d0 = xc.x - xp.x, d1 = xc.y - xp.y;
            const float d2 = xc.z - xp.z, d3 = xc.w - xp.w;
            const int o0 = row * 16 + (kq >> 1);
            // A = xx + (x - xx) * mix  for each of the three projections
            Ab[o0]            = pack_bf2(xp.x + d0 * x0.x, xp.y + d1 * x0.y);
            Ab[o0 + 1]        = pack_bf2(xp.z + d2 * x0.z, xp.w + d3 * x0.w);
            Ab[o0 + 2048]     = pack_bf2(xp.x + d0 * x1.x, xp.y + d1 * x1.y);
            Ab[o0 + 2048 + 1] = pack_bf2(xp.z + d2 * x1.z, xp.w + d3 * x1.w);
            Ab[o0 + 4096]     = pack_bf2(xp.x + d0 * x2.x, xp.y + d1 * x2.y);
            Ab[o0 + 4096 + 1] = pack_bf2(xp.z + d2 * x2.z, xp.w + d3 * x2.w);
        }
    };

    // ---- pipeline prologue ----------------------------------------------
    stage_B(0, 0);
    stage_A(0, 0);
#if HAVE_TDM
    if (wave == 0) __builtin_amdgcn_s_wait_tensorcnt(0);
#endif
    __syncthreads();

    for (int i = 0; i < NKSTEP; ++i) {
        const int cur = i & 1;
        const int nxt = cur ^ 1;
        const int k0n = (i + 1) * 32;

        if (i + 1 < NKSTEP) stage_B(nxt, k0n);   // async TDM overlaps compute

        // ---- compute on current buffer ----
        const unsigned* Bb = su + cur * KVR_BUF_U32;
        const unsigned* Ab = Bb + 3072;
#pragma unroll
        for (int p = 0; p < 3; ++p) {
            FragAB afr[2], bfr[2];
#pragma unroll
            for (int mi = 0; mi < 2; ++mi) {
                const int rbase = p * 2048 + (waveM + mi * 16 + m16) * 16;
#pragma unroll
                for (int q = 0; q < 8; ++q) {
                    // A: VGPR q holds K pair 2q + g*16
                    const int pidx = q + (g << 3);
                    afr[mi].u[q] = Ab[rbase + pidx];
                }
            }
#pragma unroll
            for (int ni = 0; ni < 2; ++ni) {
                const int rbase = p * 1024 + (waveN + ni * 16 + m16) * 16;
#pragma unroll
                for (int q = 0; q < 8; ++q)   // B: VGPR q holds K pair 2q + g*16
                    bfr[ni].u[q] = Bb[rbase + q + (g << 3)];
            }
#pragma unroll
            for (int mi = 0; mi < 2; ++mi)
#pragma unroll
                for (int ni = 0; ni < 2; ++ni)
                    acc[p][mi][ni] = __builtin_amdgcn_wmma_f32_16x16x32_bf16(
                        false, afr[mi].v, false, bfr[ni].v,
                        (short)0, acc[p][mi][ni], false, false);
        }

        if (i + 1 < NKSTEP) stage_A(nxt, k0n);
#if HAVE_TDM
        if (wave == 0) __builtin_amdgcn_s_wait_tensorcnt(0);
#endif
        __syncthreads();
    }

    // ---- epilogue: C/D layout: VGPR j -> row j + 8*g, lane -> col ----
    float* outs[3] = {ko, vo, ro};
#pragma unroll
    for (int p = 0; p < 3; ++p) {
#pragma unroll
        for (int mi = 0; mi < 2; ++mi) {
#pragma unroll
            for (int ni = 0; ni < 2; ++ni) {
                const int col = n0 + waveN + ni * 16 + m16;
#pragma unroll
                for (int j = 0; j < 8; ++j) {
                    const int row = m0 + waveM + mi * 16 + j + 8 * g;
                    outs[p][(size_t)row * C_DIM + col] = acc[p][mi][ni][j];
                }
            }
        }
    }
}

// ---------------------------------------------------------------------------
// Output projection GEMM: out[m,n] = sum_k A[m,k] * W[n,k], A pre-packed bf16.
// Single-buffered 128x128 tile, K-step 32, TDM-staged weight tile.
// Dynamic LDS: Bs = bytes [0, 8K), As = bytes [8K, 16K).
// ---------------------------------------------------------------------------
__global__ void gemm_out_kernel(const unsigned short* __restrict__ Abf, // (M,C) bf16
                                const unsigned short* __restrict__ W,   // (C,C) bf16
                                float* __restrict__ out) {              // (M,C) f32
    extern __shared__ char smem[];
    unsigned* Bs = (unsigned*)smem;
    unsigned* As = (unsigned*)(smem + 8192);

    const int tid  = threadIdx.x;
    const int n0   = blockIdx.x * 128;
    const int m0   = blockIdx.y * 128;
    const int lane = tid & 31;
    const int wave = tid >> 5;
    const int waveM = (wave & 3) << 5;
    const int waveN = (wave >> 2) << 6;
    const int m16  = lane & 15;
    const int g    = lane >> 4;

    v8f acc[2][4];
#pragma unroll
    for (int i = 0; i < 2; ++i)
#pragma unroll
        for (int j = 0; j < 4; ++j) acc[i][j] = {};

    for (int k0 = 0; k0 < C_DIM; k0 += 32) {
#if HAVE_TDM
        if (wave == 0)
            tdm_load_tile(W + (size_t)n0 * C_DIM + k0, 0u, 128);
#else
        {
            const int row  = tid >> 1;
            const int half = tid & 1;
            const uint4* src = (const uint4*)(W + (size_t)(n0 + row) * C_DIM + k0);
            uint4* dst = (uint4*)Bs + row * 4;
            dst[half]     = src[half];
            dst[half + 2] = src[half + 2];
        }
#endif
        {
            const int row  = tid >> 1;
            const int half = tid & 1;
            const uint4* src = (const uint4*)(Abf + (size_t)(m0 + row) * C_DIM + k0);
            uint4* dst = (uint4*)As + row * 4;
            dst[half]     = src[half];
            dst[half + 2] = src[half + 2];
        }
#if HAVE_TDM
        if (wave == 0) __builtin_amdgcn_s_wait_tensorcnt(0);
#endif
        __syncthreads();

        FragAB afr[2], bfr[4];
#pragma unroll
        for (int mi = 0; mi < 2; ++mi) {
            const int rbase = (waveM + mi * 16 + m16) * 16;
#pragma unroll
            for (int q = 0; q < 8; ++q) {
                const int pidx = q + (g << 3);
                afr[mi].u[q] = As[rbase + pidx];
            }
        }
#pragma unroll
        for (int ni = 0; ni < 4; ++ni) {
            const int rbase = (waveN + ni * 16 + m16) * 16;
#pragma unroll
            for (int q = 0; q < 8; ++q)
                bfr[ni].u[q] = Bs[rbase + q + (g << 3)];
        }
#pragma unroll
        for (int mi = 0; mi < 2; ++mi)
#pragma unroll
            for (int ni = 0; ni < 4; ++ni)
                acc[mi][ni] = __builtin_amdgcn_wmma_f32_16x16x32_bf16(
                    false, afr[mi].v, false, bfr[ni].v,
                    (short)0, acc[mi][ni], false, false);

        __syncthreads();
    }

#pragma unroll
    for (int mi = 0; mi < 2; ++mi) {
#pragma unroll
        for (int ni = 0; ni < 4; ++ni) {
            const int col = n0 + waveN + ni * 16 + m16;
#pragma unroll
            for (int j = 0; j < 8; ++j) {
                const int row = m0 + waveM + mi * 16 + j + 8 * g;
                out[(size_t)row * C_DIM + col] = acc[mi][ni][j];
            }
        }
    }
}

// ---------------------------------------------------------------------------
// Chunked parallel WKV scan. State (m, n, q) represents numerator N = n*e^m,
// denominator Q = q*e^m; per-step update is an associative log-sum-exp op.
// ---------------------------------------------------------------------------
__global__ void wkv_pass1(const float* __restrict__ kbuf, const float* __restrict__ vbuf,
                          const float* __restrict__ time_decay,
                          float* __restrict__ sm, float* __restrict__ sn,
                          float* __restrict__ sq) {
    const int idx = blockIdx.x * blockDim.x + threadIdx.x;
    const int c = idx % C_DIM;
    const int j = (idx / C_DIM) % NCHUNK;
    const int b = idx / (C_DIM * NCHUNK);
    const float w = -__expf(time_decay[c]);
    const size_t base = ((size_t)b * T_DIM + j * LCHUNK) * C_DIM + c;

    float m = -1e38f, n = 0.f, q = 0.f;
    for (int i = 0; i < LCHUNK; ++i) {
        const float kt = kbuf[base + (size_t)i * C_DIM];
        const float vt = vbuf[base + (size_t)i * C_DIM];
        const float md = m + w;
        const float nm = fmaxf(md, kt);
        const float ea = __expf(md - nm);
        const float eb = __expf(kt - nm);
        n = ea * n + eb * vt;
        q = ea * q + eb;
        m = nm;
    }
    sm[idx] = m; sn[idx] = n; sq[idx] = q;
}

__global__ void wkv_prefix(const float* __restrict__ sm, const float* __restrict__ sn,
                           const float* __restrict__ sq,
                           const float* __restrict__ time_decay,
                           float* __restrict__ pm, float* __restrict__ pn,
                           float* __restrict__ pq) {
    const int idx = blockIdx.x * blockDim.x + threadIdx.x;  // 0..B*C-1
    const int c = idx % C_DIM;
    const int b = idx / C_DIM;
    const float w = -__expf(time_decay[c]);
    const float Lw = (float)LCHUNK * w;

    float M = -1e38f, N = 0.f, Q = 0.f;
    for (int j = 0; j < NCHUNK; ++j) {
        const size_t o = ((size_t)b * NCHUNK + j) * C_DIM + c;
        pm[o] = M; pn[o] = N; pq[o] = Q;    // state entering chunk j
        const float m = sm[o], n = sn[o], q = sq[o];
        const float Md = M + Lw;
        const float nm = fmaxf(Md, m);
        const float ea = __expf(Md - nm);
        const float eb = __expf(m - nm);
        N = ea * N + eb * n;
        Q = ea * Q + eb * q;
        M = nm;
    }
}

__global__ void wkv_pass2(const float* __restrict__ kbuf, const float* __restrict__ vbuf,
                          const float* __restrict__ rbuf,
                          const float* __restrict__ time_decay,
                          const float* __restrict__ time_first,
                          const float* __restrict__ pm, const float* __restrict__ pn,
                          const float* __restrict__ pq,
                          unsigned short* __restrict__ a_out) {
    const int idx = blockIdx.x * blockDim.x + threadIdx.x;
    const int c = idx % C_DIM;
    const int j = (idx / C_DIM) % NCHUNK;
    const int b = idx / (C_DIM * NCHUNK);
    const float w = -__expf(time_decay[c]);
    const float u = time_first[c];
    const size_t base = ((size_t)b * T_DIM + j * LCHUNK) * C_DIM + c;

    float o = pm[idx], p = pn[idx], q = pq[idx];
    for (int i = 0; i < LCHUNK; ++i) {
        const size_t off = base + (size_t)i * C_DIM;
        const float kt = kbuf[off];
        const float vt = vbuf[off];
        const float no = fmaxf(o, u + kt);
        const float A  = __expf(o - no);
        const float Bc = __expf(u + kt - no);
        const float y  = (A * p + Bc * vt) / (A * q + Bc);
        const float no2 = fmaxf(w + o, kt);
        const float A2 = __expf(w + o - no2);
        const float B2 = __expf(kt - no2);
        p = A2 * p + B2 * vt;
        q = A2 * q + B2;
        o = no2;
        const float rt = rbuf[off];
        const float sr = 1.f / (1.f + __expf(-rt));
        const __bf16 res = (__bf16)(sr * y);
        a_out[off] = __builtin_bit_cast(unsigned short, res);
    }
}

// ---------------------------------------------------------------------------
extern "C" void kernel_launch(void* const* d_in, const int* in_sizes, int n_in,
                              void* d_out, int out_size, void* d_ws, size_t ws_size,
                              hipStream_t stream) {
    const float* x          = (const float*)d_in[0];
    const float* time_decay = (const float*)d_in[1];
    const float* time_first = (const float*)d_in[2];
    const float* tmk        = (const float*)d_in[3];
    const float* tmv        = (const float*)d_in[4];
    const float* tmr        = (const float*)d_in[5];
    const float* Wk         = (const float*)d_in[6];
    const float* Wv         = (const float*)d_in[7];
    const float* Wr         = (const float*)d_in[8];
    const float* Wo         = (const float*)d_in[9];

    // workspace layout
    char* ws = (char*)d_ws;
    const size_t wmat = (size_t)C_DIM * C_DIM;
    unsigned short* Wbf = (unsigned short*)ws;                    // 4 x C*C bf16
    float* kbuf = (float*)(ws + 4 * wmat * sizeof(unsigned short));
    float* vbuf = kbuf + (size_t)M_DIM * C_DIM;
    float* rbuf = vbuf + (size_t)M_DIM * C_DIM;
    unsigned short* abuf = (unsigned short*)(rbuf + (size_t)M_DIM * C_DIM);
    const size_t nsum = (size_t)B_DIM * NCHUNK * C_DIM;
    float* sm = (float*)(abuf + (size_t)M_DIM * C_DIM);
    float* sn = sm + nsum;
    float* sq = sn + nsum;
    float* pm = sq + nsum;
    float* pn = pm + nsum;
    float* pq = pn + nsum;

    // 1) weights -> bf16
    cvt_w_kernel<<<dim3((wmat / 4) / 256, 4), 256, 0, stream>>>(Wk, Wv, Wr, Wo, Wbf);

    // 2) fused k/v/r projections (x read once; 72 KB dynamic LDS, double-buffered)
    const dim3 kvrgrid(C_DIM / 64, M_DIM / 128);
    gemm_kvr_kernel<<<kvrgrid, 256, 2 * KVR_BUF_BYTES, stream>>>(
        x, tmk, tmv, tmr,
        Wbf + 0 * wmat, Wbf + 1 * wmat, Wbf + 2 * wmat,
        kbuf, vbuf, rbuf);

    // 3) chunked WKV scan + fused sigmoid(r)*wkv -> bf16
    const int nthr = B_DIM * NCHUNK * C_DIM;  // 262144
    wkv_pass1<<<nthr / 256, 256, 0, stream>>>(kbuf, vbuf, time_decay, sm, sn, sq);
    wkv_prefix<<<(B_DIM * C_DIM) / 256, 256, 0, stream>>>(sm, sn, sq, time_decay,
                                                          pm, pn, pq);
    wkv_pass2<<<nthr / 256, 256, 0, stream>>>(kbuf, vbuf, rbuf, time_decay, time_first,
                                              pm, pn, pq, abuf);

    // 4) output projection -> d_out (fp32, 16 KB dynamic LDS)
    gemm_out_kernel<<<dim3(C_DIM / 128, M_DIM / 128), 256, 16384, stream>>>(
        abuf, Wbf + 3 * wmat, (float*)d_out);
}
